// DAEncoder_51427938403105
// MI455X (gfx1250) — matrix operand order repaired and checked by reading, not compiled, not run on hardware
//
#include <hip/hip_runtime.h>
#include <hip/hip_bf16.h>

#define BB 256
#define SS 512
#define DD 128
#define HH 256
#define H2 512
#define GG 1024

typedef __attribute__((ext_vector_type(16))) __bf16 v16bf;
typedef __attribute__((ext_vector_type(8)))  __bf16 v8bf;
typedef __attribute__((ext_vector_type(8)))  float  v8f;

__device__ __forceinline__ __bf16 f2bf(float f) {
  unsigned u = __builtin_bit_cast(unsigned, f);
  unsigned r = u + 0x7FFFu + ((u >> 16) & 1u);
  unsigned short h = (unsigned short)(r >> 16);
  return __builtin_bit_cast(__bf16, h);
}

// Native CDNA5 V_TANH_F32 if the toolchain exposes it; libm fallback otherwise.
__device__ __forceinline__ float fast_tanh(float x) {
#if __has_builtin(__builtin_amdgcn_tanhf)
  return __builtin_amdgcn_tanhf(x);
#elif __has_builtin(__builtin_amdgcn_tanh_f32)
  return __builtin_amdgcn_tanh_f32(x);
#else
  return tanhf(x);
#endif
}

__device__ __forceinline__ float sigf(float x) { return 1.0f / (1.0f + __expf(-x)); }

// A-matrix 16x32 bf16 fragment (ISA 7.12.2). Per lane (m = lane&15, half =
// lane>>4) the 16 elements are two contiguous 8-bf16 runs of the row.
__device__ __forceinline__ v16bf ld_a_row(const __bf16* __restrict__ row, int k0,
                                          int half) {
  v8bf a0 = *(const v8bf*)(row + k0 + 8 * half);
  v8bf a1 = *(const v8bf*)(row + k0 + 16 + 8 * half);
  return __builtin_shufflevector(a0, a1, 0, 1, 2, 3, 4, 5, 6, 7, 8, 9, 10, 11, 12,
                                 13, 14, 15);
}

// B-matrix 32x16 bf16 fragment: per lane two contiguous 8-bf16 runs (same k order as A).
__device__ __forceinline__ v16bf ld_b_row(const __bf16* __restrict__ row, int k0,
                                          int half) {
  v8bf b0 = *(const v8bf*)(row + k0 + 8 * half);
  v8bf b1 = *(const v8bf*)(row + k0 + 16 + 8 * half);
  return __builtin_shufflevector(b0, b1, 0, 1, 2, 3, 4, 5, 6, 7, 8, 9, 10, 11, 12,
                                 13, 14, 15);
}

__device__ __forceinline__ v8f wmma_bf16(v16bf a, v16bf b, v8f c) {
  return __builtin_amdgcn_wmma_f32_16x16x32_bf16(false, a, false, b, (short)0, c,
                                                 false, false);
}

__device__ __forceinline__ v8f vzero8() {
  v8f z;
#pragma unroll
  for (int i = 0; i < 8; ++i) z[i] = 0.0f;
  return z;
}

// ---------------------------------------------------------------------------
// Software-pipelined 16x128 GEMM strip: double-buffered fragment sets so each
// WMMA group waits on loads issued one full stage (8 WMMAs) earlier.
// K: reduction length (multiple of 64). BS: stride (elements) between B rows.
// ---------------------------------------------------------------------------
template <int K, int BS>
__device__ __forceinline__ void gemm_pipe(const __bf16* __restrict__ arow,
                                          const __bf16* __restrict__ brow0,
                                          int half, v8f* acc) {
  v16bf a0, a1;
  v16bf b0[8], b1[8];
  a0 = ld_a_row(arow, 0, half);
#pragma unroll
  for (int nt = 0; nt < 8; ++nt)
    b0[nt] = ld_b_row(brow0 + (size_t)nt * 16 * BS, 0, half);
#pragma unroll
  for (int k0 = 0; k0 < K; k0 += 64) {
    if (k0 + 32 < K) {
      a1 = ld_a_row(arow, k0 + 32, half);
#pragma unroll
      for (int nt = 0; nt < 8; ++nt)
        b1[nt] = ld_b_row(brow0 + (size_t)nt * 16 * BS, k0 + 32, half);
    }
#pragma unroll
    for (int nt = 0; nt < 8; ++nt) acc[nt] = wmma_bf16(a0, b0[nt], acc[nt]);
    if (k0 + 64 < K) {
      a0 = ld_a_row(arow, k0 + 64, half);
#pragma unroll
      for (int nt = 0; nt < 8; ++nt)
        b0[nt] = ld_b_row(brow0 + (size_t)nt * 16 * BS, k0 + 64, half);
    }
    if (k0 + 32 < K) {
#pragma unroll
      for (int nt = 0; nt < 8; ++nt) acc[nt] = wmma_bf16(a1, b1[nt], acc[nt]);
    }
  }
}

// ---------------------------------------------------------------------------
// Setup: bf16 conversions (one-time).
// ---------------------------------------------------------------------------
__global__ __launch_bounds__(256) void k_cvt(const float* __restrict__ src,
                                             __bf16* __restrict__ dst, int n) {
  int i = blockIdx.x * 256 + threadIdx.x;
  if (i < n) dst[i] = f2bf(src[i]);
}

// Split W_ue (512x1024) into W_h (cols 0..511) and W_x (cols 512..1023), bf16.
__global__ __launch_bounds__(256) void k_cvt_wue(const float* __restrict__ W_ue,
                                                 __bf16* __restrict__ Wh,
                                                 __bf16* __restrict__ Wx) {
  int i = blockIdx.x * 256 + threadIdx.x;  // over 512*1024
  int o = i >> 10, col = i & 1023;
  float v = W_ue[i];
  if (col < 512)
    Wh[(size_t)o * 512 + col] = f2bf(v);
  else
    Wx[(size_t)o * 512 + (col - 512)] = f2bf(v);
}

// Transpose-convert x (B,S,D) f32 -> xbt (B,D,S) bf16 via LDS 32x32 tiles.
__global__ __launch_bounds__(256) void k_xt(const float* __restrict__ x,
                                            __bf16* __restrict__ xbt) {
  __shared__ float tile[32][33];
  int b = blockIdx.y;
  int s0 = (blockIdx.x & 15) * 32;
  int d0 = (blockIdx.x >> 4) * 32;
  int tx = threadIdx.x & 31, ty = threadIdx.x >> 5;  // 32 x 8
#pragma unroll
  for (int j = 0; j < 4; ++j) {
    int s = ty + j * 8;
    tile[s][tx] = x[((size_t)b * SS + (s0 + s)) * DD + (d0 + tx)];
  }
  __syncthreads();
#pragma unroll
  for (int j = 0; j < 4; ++j) {
    int d_ = ty + j * 8;
    xbt[((size_t)b * DD + (d0 + d_)) * SS + (s0 + tx)] = f2bf(tile[tx][d_]);
  }
}

__global__ __launch_bounds__(256) void k_init(float* __restrict__ c,
                                              __bf16* __restrict__ hs) {
  int idx = blockIdx.x * 256 + threadIdx.x;
  if (idx < BB * HH) c[idx] = 0.0f;
  if (idx < BB * H2) hs[idx] = f2bf(0.0f);
}

// ---------------------------------------------------------------------------
// K1: px[b][o][d] = sum_s xbt[b,d,s] * W_x[o,s]
// 8192 wave-tasks: b(256) x d-tile(8) x o-group(4); each wave does 16x128.
// ---------------------------------------------------------------------------
__global__ __launch_bounds__(128) void k_proj_x(const __bf16* __restrict__ xbt,
                                                const __bf16* __restrict__ Wx,
                                                float* __restrict__ px) {
  int wave = blockIdx.x * 4 + (threadIdx.x >> 5);
  int lane = threadIdx.x & 31;
  int ng = wave & 3; wave >>= 2;
  int mt = wave & 7; wave >>= 3;
  int b = wave;
  int m0 = mt * 16;   // d
  int n0 = ng * 128;  // o
  int l15 = lane & 15, half = lane >> 4;
  const __bf16* arow = xbt + ((size_t)b * DD + (m0 + l15)) * SS;
  const __bf16* brow0 = Wx + (size_t)(n0 + l15) * SS;
  v8f acc[8];
#pragma unroll
  for (int i = 0; i < 8; ++i) acc[i] = vzero8();

  gemm_pipe<SS, SS>(arow, brow0, half, acc);

#pragma unroll
  for (int nt = 0; nt < 8; ++nt) {
#pragma unroll
    for (int r = 0; r < 8; ++r) {
      int d_ = m0 + r + 8 * half;
      int o_ = n0 + nt * 16 + l15;
      px[(size_t)b * SS * DD + (size_t)o_ * DD + d_] = acc[nt][r];
    }
  }
}

// ---------------------------------------------------------------------------
// K2: ph[b][o] = sum_j hs_bf[b][j] * W_h[o][j]
// 64 wave-tasks: b-tile(16) x o-group(4); each wave 16x128, K=512.
// ---------------------------------------------------------------------------
__global__ __launch_bounds__(128) void k_proj_h(const __bf16* __restrict__ hs,
                                                const __bf16* __restrict__ Wh,
                                                float* __restrict__ ph) {
  int wave = blockIdx.x * 4 + (threadIdx.x >> 5);
  int lane = threadIdx.x & 31;
  int ng = wave & 3;
  int mt = wave >> 2;
  int m0 = mt * 16;   // b
  int n0 = ng * 128;  // o
  int l15 = lane & 15, half = lane >> 4;
  const __bf16* arow = hs + (size_t)(m0 + l15) * H2;
  const __bf16* brow0 = Wh + (size_t)(n0 + l15) * H2;
  v8f acc[8];
#pragma unroll
  for (int i = 0; i < 8; ++i) acc[i] = vzero8();

  gemm_pipe<H2, H2>(arow, brow0, half, acc);

#pragma unroll
  for (int nt = 0; nt < 8; ++nt) {
#pragma unroll
    for (int r = 0; r < 8; ++r) {
      int b_ = m0 + r + 8 * half;
      int o_ = n0 + nt * 16 + l15;
      ph[(size_t)b_ * SS + o_] = acc[nt][r];
    }
  }
}

// ---------------------------------------------------------------------------
// K3: E[b,d] = sum_o tanh(ph[b,o] + px[b,o,d]) * v[o]; softmax over d;
//     inp_bf[b,d] = bf16(x[b,t,d] * alpha[b,d]).  One block per b, 128 thr (=D).
//     px reads are coalesced and (64 MB) L2-resident across all 512 steps.
// ---------------------------------------------------------------------------
__global__ __launch_bounds__(128) void k_attn(const float* __restrict__ ph,
                                              const float* __restrict__ px,
                                              const float* __restrict__ v_e,
                                              const float* __restrict__ x,
                                              __bf16* __restrict__ inp_bf, int t) {
  int b = blockIdx.x;
  int d = threadIdx.x;
  __shared__ float sph[SS];
  __shared__ float sv[SS];
  __shared__ float red[DD];
  for (int i = d; i < SS; i += DD) {
    sph[i] = ph[(size_t)b * SS + i];
    sv[i] = v_e[i];
  }
  __syncthreads();
  const float* pxb = px + (size_t)b * SS * DD + d;
  float e = 0.0f;
#pragma unroll 8
  for (int o = 0; o < SS; ++o) {
    e += fast_tanh(sph[o] + pxb[(size_t)o * DD]) * sv[o];
  }
  red[d] = e;
  __syncthreads();
  for (int s2 = 64; s2 > 0; s2 >>= 1) {
    if (d < s2) red[d] = fmaxf(red[d], red[d + s2]);
    __syncthreads();
  }
  float mx = red[0];
  __syncthreads();
  float ex = __expf(e - mx);
  red[d] = ex;
  __syncthreads();
  for (int s2 = 64; s2 > 0; s2 >>= 1) {
    if (d < s2) red[d] += red[d + s2];
    __syncthreads();
  }
  float alpha = ex / red[0];
  float inp = x[((size_t)b * SS + t) * DD + d] * alpha;
  inp_bf[(size_t)b * DD + d] = f2bf(inp);
}

// ---------------------------------------------------------------------------
// K4: gates[b][g] = inp@W_ih^T + h@W_hh^T  (biases added in K5)
// 128 wave-tasks: b-tile(16) x g-group(8); each wave 16x128, K = 128 + 256.
// ---------------------------------------------------------------------------
__global__ __launch_bounds__(128) void k_gates(const __bf16* __restrict__ inp_bf,
                                               const __bf16* __restrict__ hs,
                                               const __bf16* __restrict__ Wih,
                                               const __bf16* __restrict__ Whh,
                                               float* __restrict__ gates) {
  int wave = blockIdx.x * 4 + (threadIdx.x >> 5);
  int lane = threadIdx.x & 31;
  int ng = wave & 7;
  int mt = wave >> 3;
  int m0 = mt * 16;   // b
  int n0 = ng * 128;  // g
  int l15 = lane & 15, half = lane >> 4;
  v8f acc[8];
#pragma unroll
  for (int i = 0; i < 8; ++i) acc[i] = vzero8();

  // segment 1: K = 128 over inp / W_ih
  const __bf16* arow1 = inp_bf + (size_t)(m0 + l15) * DD;
  const __bf16* brow1 = Wih + (size_t)(n0 + l15) * DD;
  gemm_pipe<DD, DD>(arow1, brow1, half, acc);

  // segment 2: K = 256 over h (= first half of hs) / W_hh
  const __bf16* arow2 = hs + (size_t)(m0 + l15) * H2;
  const __bf16* brow2 = Whh + (size_t)(n0 + l15) * HH;
  gemm_pipe<HH, HH>(arow2, brow2, half, acc);

#pragma unroll
  for (int nt = 0; nt < 8; ++nt) {
#pragma unroll
    for (int r = 0; r < 8; ++r) {
      int b_ = m0 + r + 8 * half;
      int g_ = n0 + nt * 16 + l15;
      gates[(size_t)b_ * GG + g_] = acc[nt][r];
    }
  }
}

// ---------------------------------------------------------------------------
// K5: LSTM cell elementwise; writes c (f32), [h|c] (bf16) for next step,
//     and h_new to out[t][b][h].
// ---------------------------------------------------------------------------
__global__ __launch_bounds__(256) void k_lstm(const float* __restrict__ gates,
                                              const float* __restrict__ b_ih,
                                              const float* __restrict__ b_hh,
                                              float* __restrict__ c,
                                              __bf16* __restrict__ hs,
                                              float* __restrict__ out, int t) {
  int idx = blockIdx.x * 256 + threadIdx.x;  // over B*H
  int b = idx >> 8;
  int h_ = idx & 255;
  const float* g = gates + (size_t)b * GG;
  float iv = sigf(g[h_] + b_ih[h_] + b_hh[h_]);
  float fv = sigf(g[HH + h_] + b_ih[HH + h_] + b_hh[HH + h_]);
  float gv = fast_tanh(g[2 * HH + h_] + b_ih[2 * HH + h_] + b_hh[2 * HH + h_]);
  float ov = sigf(g[3 * HH + h_] + b_ih[3 * HH + h_] + b_hh[3 * HH + h_]);
  float cn = fv * c[idx] + iv * gv;
  float hn = ov * fast_tanh(cn);
  c[idx] = cn;
  hs[(size_t)b * H2 + h_] = f2bf(hn);
  hs[(size_t)b * H2 + HH + h_] = f2bf(cn);
  out[((size_t)t * BB + b) * HH + h_] = hn;
}

extern "C" void kernel_launch(void* const* d_in, const int* in_sizes, int n_in,
                              void* d_out, int out_size, void* d_ws, size_t ws_size,
                              hipStream_t stream) {
  const float* x    = (const float*)d_in[0];  // (B,S,D)
  const float* W_ue = (const float*)d_in[1];  // (S, 2H+S) = (512,1024)
  const float* v_e  = (const float*)d_in[2];  // (1,S)
  const float* W_ih = (const float*)d_in[3];  // (4H,D)
  const float* W_hh = (const float*)d_in[4];  // (4H,H)
  const float* b_ih = (const float*)d_in[5];  // (4H)
  const float* b_hh = (const float*)d_in[6];  // (4H)
  float* out = (float*)d_out;                 // (S,B,H)

  unsigned char* w = (unsigned char*)d_ws;
  size_t off = 0;
  float* px = (float*)(w + off);        off += (size_t)BB * SS * DD * 4;  // 64 MB
  __bf16* xbt = (__bf16*)(w + off);     off += (size_t)BB * DD * SS * 2;  // 32 MB
  __bf16* Wh_bf = (__bf16*)(w + off);   off += (size_t)SS * H2 * 2;       // 512 KB
  __bf16* Wx_bf = (__bf16*)(w + off);   off += (size_t)SS * SS * 2;       // 512 KB
  __bf16* Wih_bf = (__bf16*)(w + off);  off += (size_t)GG * DD * 2;       // 256 KB
  __bf16* Whh_bf = (__bf16*)(w + off);  off += (size_t)GG * HH * 2;       // 512 KB
  float* ph = (float*)(w + off);        off += (size_t)BB * SS * 4;       // 512 KB
  float* gates = (float*)(w + off);     off += (size_t)BB * GG * 4;       // 1 MB
  float* c = (float*)(w + off);         off += (size_t)BB * HH * 4;       // 256 KB
  __bf16* hs = (__bf16*)(w + off);      off += (size_t)BB * H2 * 2;       // 256 KB
  __bf16* inp_bf = (__bf16*)(w + off);  off += (size_t)BB * DD * 2;       // 64 KB

  // one-time setup
  k_init<<<512, 256, 0, stream>>>(c, hs);
  k_cvt_wue<<<2048, 256, 0, stream>>>(W_ue, Wh_bf, Wx_bf);
  k_cvt<<<512, 256, 0, stream>>>(W_ih, Wih_bf, GG * DD);
  k_cvt<<<1024, 256, 0, stream>>>(W_hh, Whh_bf, GG * HH);
  {
    dim3 g(64, BB);
    k_xt<<<g, 256, 0, stream>>>(x, xbt);
  }
  k_proj_x<<<2048, 128, 0, stream>>>(xbt, Wx_bf, px);

  for (int t = 0; t < SS; ++t) {
    k_proj_h<<<16, 128, 0, stream>>>(hs, Wh_bf, ph);
    k_attn<<<256, 128, 0, stream>>>(ph, px, v_e, x, inp_bf, t);
    k_gates<<<32, 128, 0, stream>>>(inp_bf, hs, Wih_bf, Whh_bf, gates);
    k_lstm<<<256, 256, 0, stream>>>(gates, b_ih, b_hh, c, hs, out, t);
  }
}
